// MultiHeadAttention_730144440560
// MI455X (gfx1250) — matrix operand order repaired and checked by reading, not compiled, not run on hardware
//
#include <hip/hip_runtime.h>
#include <stdint.h>

typedef __attribute__((ext_vector_type(16))) __bf16 v16bf;
typedef __attribute__((ext_vector_type(8)))  float  v8f;

union FragAB { v16bf bf; uint32_t u[8]; };
union FragC  { v8f f; float e[8]; };

__device__ __forceinline__ uint16_t f2bf(float x) {
  uint32_t u = __float_as_uint(x);
  uint32_t r = (u + 0x7FFFu + ((u >> 16) & 1u)) >> 16;
  return (uint16_t)r;
}

__device__ __forceinline__ v8f zero_v8f() {
  FragC c;
#pragma unroll
  for (int i = 0; i < 8; ++i) c.e[i] = 0.0f;
  return c.f;
}

// A-fragment (16x32, bf16) from row-major src (leading dim ld, elements).
// ISA 16-bit A layout: lanes 0-15 -> K {0..7,16..23}, lanes 16-31 -> K {8..15,24..31}
__device__ __forceinline__ void loadA_g(const uint16_t* __restrict__ src, int ld,
                                        int m0, int k0, int lane, FragAB& a) {
  const int m  = m0 + (lane & 15);
  const int h8 = (lane >> 4) * 8;
  const uint16_t* p = src + (size_t)m * ld + k0 + h8;
#pragma unroll
  for (int v = 0; v < 4; ++v) a.u[v]     = *(const uint32_t*)(p + 2 * v);
#pragma unroll
  for (int v = 0; v < 4; ++v) a.u[4 + v] = *(const uint32_t*)(p + 16 + 2 * v);
}

// B-fragment (32x16, bf16) from B^T stored row-major (N x K, leading dim ld).
// ISA 16-bit B layout: lanes 0-15 -> K {0..7,16..23}, lanes 16-31 -> K {8..15,24..31}
__device__ __forceinline__ void loadBT_g(const uint16_t* __restrict__ bt, int ld,
                                         int n0, int k0, int lane, FragAB& b) {
  const int n  = n0 + (lane & 15);
  const int kh = k0 + (lane >> 4) * 8;
  const uint16_t* p = bt + (size_t)n * ld + kh;
#pragma unroll
  for (int v = 0; v < 8; ++v) b.u[v] = *(const uint32_t*)(p + 2 * v + (v >= 4 ? 8 : 0));
}

__device__ __forceinline__ v8f wmma_bf16(const FragAB& a, const FragAB& b, v8f c) {
  return __builtin_amdgcn_wmma_f32_16x16x32_bf16(false, a.bf, false, b.bf,
                                                 (short)0, c, false, false);
}

// ---------------- conversion / transpose ----------------

__global__ void __launch_bounds__(256) k_cvt_bf16(const float* __restrict__ src,
                                                  uint16_t* __restrict__ dst, int n) {
  int i = blockIdx.x * 256 + threadIdx.x;
  if (i < n) dst[i] = f2bf(src[i]);
}

// src: K x N row-major f32 ; dst: N x K row-major bf16 (i.e. transposed)
__global__ void __launch_bounds__(256) k_transpose_bf16(const float* __restrict__ src,
                                                        uint16_t* __restrict__ dst,
                                                        int K, int N) {
  int i = blockIdx.x * 256 + threadIdx.x;
  if (i < K * N) {
    int n = i / K, k = i % K;
    dst[i] = f2bf(src[(size_t)k * N + n]);
  }
}

// ---------------- QKV projection GEMM ----------------
// xb: 4096x768 bf16 ; wT: 2304x768 bf16 (= w_qkv^T) ; bias: 2304 f32
// Scatter epilogue: col c = (head*64+dd)*3 + which
//   Q,K stored (B,h,N,d) ; V stored transposed (B,h,d,N)
__global__ void __launch_bounds__(256) k_gemm_qkv(
    const uint16_t* __restrict__ xb, const uint16_t* __restrict__ wT,
    const float* __restrict__ bias,
    uint16_t* __restrict__ Qb, uint16_t* __restrict__ Kb, uint16_t* __restrict__ Vt) {
  const int lane  = threadIdx.x & 31;
  const int wid   = blockIdx.x * 8 + (threadIdx.x >> 5);
  const int mtile = wid / 36;
  const int ngrp  = wid % 36;
  const int m0 = mtile * 16;
  const int n0 = ngrp * 64;

  FragC acc[4];
#pragma unroll
  for (int t = 0; t < 4; ++t) acc[t].f = zero_v8f();

  for (int k0 = 0; k0 < 768; k0 += 32) {
    FragAB a;
    loadA_g(xb, 768, m0, k0, lane, a);
#pragma unroll
    for (int t = 0; t < 4; ++t) {
      FragAB b;
      loadBT_g(wT, 768, n0 + 16 * t, k0, lane, b);
      acc[t].f = wmma_bf16(a, b, acc[t].f);
    }
  }

  const int rh = (lane >> 4) * 8;
  const int cl = lane & 15;
#pragma unroll
  for (int t = 0; t < 4; ++t) {
    const int c     = n0 + 16 * t + cl;
    const int which = c % 3;
    const int hd    = c / 3;          // head*64 + dd
    const int head  = hd >> 6;
    const int dd    = hd & 63;
    const float bv  = bias[c];
#pragma unroll
    for (int r = 0; r < 8; ++r) {
      const int m    = m0 + rh + r;   // global row = b*1024 + n
      const int bidx = m >> 10;
      const int nn   = m & 1023;
      const int bh   = bidx * 12 + head;
      const uint16_t hv = f2bf(acc[t].e[r] + bv);
      if (which == 0)      Qb[((size_t)bh * 1024 + nn) * 64 + dd] = hv;
      else if (which == 1) Kb[((size_t)bh * 1024 + nn) * 64 + dd] = hv;
      else                 Vt[((size_t)bh * 64 + dd) * 1024 + nn] = hv;
    }
  }
}

// ---------------- fused flash attention ----------------
// One wave per (b*h, 16-row query tile). 48 heads * 64 tiles = 3072 waves.
__global__ void __launch_bounds__(128) k_attn(
    const uint16_t* __restrict__ Qb, const uint16_t* __restrict__ Kb,
    const uint16_t* __restrict__ Vt, uint16_t* __restrict__ attnb) {
  __shared__ uint16_t pb[4][16 * 32];   // per-wave 16x32 P tile (bf16)

  const int lane = threadIdx.x & 31;
  const int w    = threadIdx.x >> 5;
  const int wid  = blockIdx.x * 4 + w;
  const int bh   = wid >> 6;
  const int q0   = (wid & 63) * 16;

  const uint16_t* Qh = Qb + (size_t)bh * 1024 * 64;
  const uint16_t* Kh = Kb + (size_t)bh * 1024 * 64;
  const uint16_t* Vh = Vt + (size_t)bh * 64 * 1024;

  FragAB aq0, aq1;
  loadA_g(Qh, 64, q0, 0,  lane, aq0);
  loadA_g(Qh, 64, q0, 32, lane, aq1);

  FragC o[4];
  float mrow[8], lrow[8];
#pragma unroll
  for (int c = 0; c < 4; ++c) o[c].f = zero_v8f();
#pragma unroll
  for (int r = 0; r < 8; ++r) { mrow[r] = -1e30f; lrow[r] = 0.0f; }

  const float scale = 0.03608439182435161f;   // 1/sqrt(768)
  const float L2E   = 1.4426950408889634f;
  const int rh = (lane >> 4) * 8;
  const int cl = lane & 15;

  for (int j0 = 0; j0 < 1024; j0 += 32) {
    // ---- scores S = Q K^T for 32 keys ----
    FragAB b;
    FragC s0, s1;
    loadBT_g(Kh, 64, j0,      0,  lane, b); s0.f = wmma_bf16(aq0, b, zero_v8f());
    loadBT_g(Kh, 64, j0,      32, lane, b); s0.f = wmma_bf16(aq1, b, s0.f);
    loadBT_g(Kh, 64, j0 + 16, 0,  lane, b); s1.f = wmma_bf16(aq0, b, zero_v8f());
    loadBT_g(Kh, 64, j0 + 16, 32, lane, b); s1.f = wmma_bf16(aq1, b, s1.f);

    // ---- flash softmax update (per row r+8*half) ----
#pragma unroll
    for (int r = 0; r < 8; ++r) {
      float t0 = s0.e[r] * scale;
      float t1 = s1.e[r] * scale;
      float mx = fmaxf(t0, t1);
      mx = fmaxf(mx, __shfl_xor(mx, 1, 32));
      mx = fmaxf(mx, __shfl_xor(mx, 2, 32));
      mx = fmaxf(mx, __shfl_xor(mx, 4, 32));
      mx = fmaxf(mx, __shfl_xor(mx, 8, 32));
      const float mn    = fmaxf(mrow[r], mx);
      const float alpha = exp2f((mrow[r] - mn) * L2E);
      mrow[r] = mn;
      const float p0 = exp2f((t0 - mn) * L2E);
      const float p1 = exp2f((t1 - mn) * L2E);
      float sum = p0 + p1;
      sum += __shfl_xor(sum, 1, 32);
      sum += __shfl_xor(sum, 2, 32);
      sum += __shfl_xor(sum, 4, 32);
      sum += __shfl_xor(sum, 8, 32);
      lrow[r] = lrow[r] * alpha + sum;
#pragma unroll
      for (int c = 0; c < 4; ++c) o[c].e[r] *= alpha;
      pb[w][(rh + r) * 32 + cl]      = f2bf(p0);
      pb[w][(rh + r) * 32 + 16 + cl] = f2bf(p1);
    }
    asm volatile("s_wait_dscnt 0" ::: "memory");

    // ---- re-layout P (C-frag -> A-frag) via LDS ----
    FragAB ap;
    {
      const int m = lane & 15;
      const uint16_t* p = &pb[w][m * 32 + rh];
#pragma unroll
      for (int v = 0; v < 4; ++v) ap.u[v]     = *(const uint32_t*)(p + 2 * v);
#pragma unroll
      for (int v = 0; v < 4; ++v) ap.u[4 + v] = *(const uint32_t*)(p + 16 + 2 * v);
    }

    // ---- O += P V  (V pre-transposed: d x N, contiguous over keys) ----
#pragma unroll
    for (int c = 0; c < 4; ++c) {
      FragAB bv;
      loadBT_g(Vh, 1024, c * 16, j0, lane, bv);
      o[c].f = wmma_bf16(ap, bv, o[c].f);
    }
  }

  // ---- normalize and store bf16 in (B, N, C) order ----
  const int bI = bh / 12, head = bh % 12;
#pragma unroll
  for (int r = 0; r < 8; ++r) {
    const float inv = 1.0f / lrow[r];
    const int n = q0 + rh + r;
#pragma unroll
    for (int c = 0; c < 4; ++c) {
      const int col = head * 64 + c * 16 + cl;
      attnb[(size_t)(bI * 1024 + n) * 768 + col] = f2bf(o[c].e[r] * inv);
    }
  }
}

// ---------------- output projection GEMM ----------------
__global__ void __launch_bounds__(256) k_gemm_out(
    const uint16_t* __restrict__ ab, const uint16_t* __restrict__ wT,
    const float* __restrict__ bias, float* __restrict__ out) {
  const int lane  = threadIdx.x & 31;
  const int wid   = blockIdx.x * 8 + (threadIdx.x >> 5);
  const int mtile = wid / 12;
  const int ngrp  = wid % 12;
  const int m0 = mtile * 16;
  const int n0 = ngrp * 64;

  FragC acc[4];
#pragma unroll
  for (int t = 0; t < 4; ++t) acc[t].f = zero_v8f();

  for (int k0 = 0; k0 < 768; k0 += 32) {
    FragAB a;
    loadA_g(ab, 768, m0, k0, lane, a);
#pragma unroll
    for (int t = 0; t < 4; ++t) {
      FragAB b;
      loadBT_g(wT, 768, n0 + 16 * t, k0, lane, b);
      acc[t].f = wmma_bf16(a, b, acc[t].f);
    }
  }

  const int rh = (lane >> 4) * 8;
  const int cl = lane & 15;
#pragma unroll
  for (int t = 0; t < 4; ++t) {
    const int c   = n0 + 16 * t + cl;
    const float bv = bias[c];
#pragma unroll
    for (int r = 0; r < 8; ++r)
      out[(size_t)(m0 + rh + r) * 768 + c] = acc[t].e[r] + bv;
  }
}

extern "C" void kernel_launch(void* const* d_in, const int* in_sizes, int n_in,
                              void* d_out, int out_size, void* d_ws, size_t ws_size,
                              hipStream_t stream) {
  (void)in_sizes; (void)n_in; (void)out_size; (void)ws_size;
  const float* x      = (const float*)d_in[0];   // (4,1024,768)
  const float* w_qkv  = (const float*)d_in[1];   // (768,2304)
  const float* b_qkv  = (const float*)d_in[2];   // (2304,)
  const float* w_proj = (const float*)d_in[3];   // (768,768)
  const float* b_proj = (const float*)d_in[4];   // (768,)

  char* ws = (char*)d_ws;
  uint16_t* xb     = (uint16_t*)(ws + 0);          // 6,291,456 B
  uint16_t* wqkvT  = (uint16_t*)(ws + 6291456);    // 3,538,944 B
  uint16_t* wprojT = (uint16_t*)(ws + 9830400);    // 1,179,648 B
  uint16_t* Qb     = (uint16_t*)(ws + 11010048);   // 6,291,456 B
  uint16_t* Kb     = (uint16_t*)(ws + 17301504);   // 6,291,456 B
  uint16_t* Vt     = (uint16_t*)(ws + 23592960);   // 6,291,456 B
  uint16_t* attnb  = (uint16_t*)(ws + 29884416);   // 6,291,456 B -> total 36,175,872 B

  k_cvt_bf16<<<(3145728 + 255) / 256, 256, 0, stream>>>(x, xb, 3145728);
  k_transpose_bf16<<<(1769472 + 255) / 256, 256, 0, stream>>>(w_qkv, wqkvT, 768, 2304);
  k_transpose_bf16<<<(589824 + 255) / 256, 256, 0, stream>>>(w_proj, wprojT, 768, 768);

  k_gemm_qkv<<<1152, 256, 0, stream>>>(xb, wqkvT, b_qkv, Qb, Kb, Vt);  // 9216 waves
  k_attn<<<768, 128, 0, stream>>>(Qb, Kb, Vt, attnb);                  // 3072 waves
  k_gemm_out<<<384, 256, 0, stream>>>(attnb, wprojT, b_proj, (float*)d_out); // 3072 waves
}
